// ScaledDotProductGeometryAttentionM_4672924418745
// MI455X (gfx1250) — compile-verified, matrix operand order checked by reading, not run on hardware
//
#include <hip/hip_runtime.h>
#include <hip/hip_bf16.h>

// Problem constants
#define B_   8
#define NQ_  1024
#define NK_  1024
#define DM_  512
#define H_   8
#define DK_  64
#define HD_  512   // H*DK

typedef __attribute__((ext_vector_type(16))) __bf16 v16bf;
typedef __attribute__((ext_vector_type(8)))  float  v8f;
typedef unsigned int __attribute__((ext_vector_type(4))) v4u;
typedef int __attribute__((ext_vector_type(8))) v8i;
typedef int __attribute__((ext_vector_type(4))) v4i;

static __device__ __forceinline__ v8f wmma_acc(v16bf a, v16bf b, v8f c) {
  // D = A(16x32 bf16) * B(32x16 bf16) + C(16x16 f32)
  return __builtin_amdgcn_wmma_f32_16x16x32_bf16(
      /*neg_a=*/false, a, /*neg_b=*/false, b,
      /*c_mod=*/(short)0, c, /*reuse_a=*/false, /*reuse_b=*/false);
}

// Load A-fragment / B^T-fragment (16 rows x 32 K) from row-major bf16:
// lane = half*16 + r ; lane holds row r, K chunks {half*8..+7} and {16+half*8..+7}
static __device__ __forceinline__ v16bf load_frag_bf16(const __bf16* base, int ld) {
  const int lane = threadIdx.x & 31;
  const int r  = lane & 15;
  const int kh = (lane >> 4) << 3;          // 0 or 8
  union { v16bf v; uint4 q[2]; } u;
  u.q[0] = *(const uint4*)(base + (size_t)r * ld + kh);
  u.q[1] = *(const uint4*)(base + (size_t)r * ld + kh + 16);
  return u.v;
}

// Same fragment but source is f32; convert to bf16 while loading.
static __device__ __forceinline__ v16bf load_frag_f32(const float* base, int ld) {
  const int lane = threadIdx.x & 31;
  const int r  = lane & 15;
  const int kh = (lane >> 4) << 3;
  union { v16bf v; __bf16 e[16]; } u;
  const float* p = base + (size_t)r * ld + kh;
#pragma unroll
  for (int i = 0; i < 8; ++i) u.e[i]     = (__bf16)p[i];
#pragma unroll
  for (int i = 0; i < 8; ++i) u.e[8 + i] = (__bf16)p[16 + i];
  return u.v;
}

// ---------------------------------------------------------------------------
// TDM: 2-D tile load (bf16 elements) Global -> LDS, per CDNA5 ISA.
// rows x row_elems tile; tensor row stride = row_stride_elems; issued per-wave,
// tracked with TENSORcnt. 6-arg builtin form (clang-23 toolchain).
// ---------------------------------------------------------------------------
static __device__ __forceinline__ void tdm_load_2d_bf16(
    unsigned lds_off, const void* gaddr, unsigned rows, unsigned row_elems,
    unsigned row_stride_elems, unsigned tensor_rows) {
  const unsigned long long ga = (unsigned long long)(uintptr_t)gaddr;
  v4u g0;
  g0[0] = 1u;                                    // count=1, user mode
  g0[1] = lds_off;                               // lds_addr (bytes)
  g0[2] = (unsigned)ga;                          // global_addr[31:0]
  g0[3] = (unsigned)((ga >> 32) & 0x01FFFFFFu)   // global_addr[56:32]
        | (2u << 30);                            // type=2 ("image")
  v8i g1;
  g1[0] = (int)(1u << 16);                       // workgroup_mask=0, data_size=1 (2B)
  g1[1] = (int)((row_elems & 0xFFFFu) << 16);    // tensor_dim0[15:0] @ bits[63:48]
  g1[2] = (int)(((row_elems >> 16) & 0xFFFFu) |  // tensor_dim0[31:16]
                ((tensor_rows & 0xFFFFu) << 16));// tensor_dim1[15:0]
  g1[3] = (int)(((tensor_rows >> 16) & 0xFFFFu) |// tensor_dim1[31:16]
                ((row_elems & 0xFFFFu) << 16));  // tile_dim0
  g1[4] = (int)(rows & 0xFFFFu);                 // tile_dim1 (tile_dim2=0)
  g1[5] = (int)row_stride_elems;                 // tensor_dim0_stride[31:0]
  g1[6] = 0;                                     // stride hi / dim1_stride (unused)
  g1[7] = 0;
  const v4i gz4 = {0, 0, 0, 0};
  const v8i gz8 = {0, 0, 0, 0, 0, 0, 0, 0};
  __builtin_amdgcn_tensor_load_to_lds(g0, g1, gz4, gz4, gz8, 0);
}

// ---------------------------------------------------------------------------
// Kernel 1: transpose+convert the four 512x512 weight matrices to bf16 W^T[j][k]
// ---------------------------------------------------------------------------
__global__ __launch_bounds__(256) void pack_weights_kernel(
    const float* __restrict__ Wq, const float* __restrict__ Wk,
    const float* __restrict__ Wv, const float* __restrict__ Wo,
    __bf16* __restrict__ WqT, __bf16* __restrict__ WkT,
    __bf16* __restrict__ WvT, __bf16* __restrict__ WoT) {
  const float* src; __bf16* dst;
  switch (blockIdx.y) {
    case 0: src = Wq; dst = WqT; break;
    case 1: src = Wk; dst = WkT; break;
    case 2: src = Wv; dst = WvT; break;
    default: src = Wo; dst = WoT; break;
  }
  const int idx = blockIdx.x * 256 + threadIdx.x;  // 0..262143
  const int j = idx >> 9;                           // output row (col of W)
  const int k = idx & 511;
  dst[idx] = (__bf16)src[(size_t)k * 512 + j];
}

// ---------------------------------------------------------------------------
// Kernel 2: fused Q/K/V projections (M=8192, N=512, K=512 bf16 WMMA GEMM)
// wave computes a 16x128 tile (8 accumulators); epilogue applies bias and the
// memory gate (memory*Wmm + bmm) for K/V; stores bf16 [b][h][n][d].
// ---------------------------------------------------------------------------
__global__ __launch_bounds__(256) void proj_kernel(
    const float* __restrict__ queries, const float* __restrict__ keys,
    const float* __restrict__ values,
    const __bf16* __restrict__ WqT, const __bf16* __restrict__ WkT,
    const __bf16* __restrict__ WvT,
    const float* __restrict__ bq, const float* __restrict__ bk,
    const float* __restrict__ bv,
    const float* __restrict__ memory, const float* __restrict__ Wmm,
    const float* __restrict__ bmm,
    __bf16* __restrict__ Qb, __bf16* __restrict__ Kg, __bf16* __restrict__ Vg) {
  const int z = blockIdx.z;
  const float*  act  = (z == 0) ? queries : (z == 1) ? keys : values;
  const __bf16* Wt   = (z == 0) ? WqT : (z == 1) ? WkT : WvT;
  const float*  bias = (z == 0) ? bq : (z == 1) ? bk : bv;
  __bf16*       dst  = (z == 0) ? Qb : (z == 1) ? Kg : Vg;
  const bool gate = (z != 0);

  const int wave = threadIdx.x >> 5;
  const int lane = threadIdx.x & 31;
  const int half = lane >> 4;
  const int cl   = lane & 15;
  const int m0 = blockIdx.x * 128 + wave * 16;
  const int n0 = blockIdx.y * 128;

  v8f acc[8] = {};
  for (int k = 0; k < DM_; k += 32) {
    v16bf a = load_frag_f32(act + (size_t)m0 * DM_ + k, DM_);
#pragma unroll
    for (int t = 0; t < 8; ++t) {
      v16bf b = load_frag_bf16(Wt + (size_t)(n0 + 16 * t) * DM_ + k, DM_);
      acc[t] = wmma_acc(a, b, acc[t]);
    }
  }

  float mem[8] = {};
  if (gate) {
#pragma unroll
    for (int g = 0; g < 8; ++g) {
      const int row = m0 + g + (half << 3);
      mem[g] = memory[((row >> 10) << 10) + (row & 1023)];
    }
  }
#pragma unroll
  for (int t = 0; t < 8; ++t) {
    const int col = n0 + t * 16 + cl;
    const float bcol = bias[col];
    const float wm = gate ? Wmm[col] : 0.f;
    const float bm = gate ? bmm[col] : 0.f;
    const int hh = col >> 6, dd = col & 63;
#pragma unroll
    for (int g = 0; g < 8; ++g) {
      const int row = m0 + g + (half << 3);
      const int bb = row >> 10, nn = row & 1023;
      float v = acc[t][g] + bcol;
      if (gate) v *= (mem[g] * wm + bm);
      dst[(((size_t)bb * H_ + hh) * NQ_ + nn) * DK_ + dd] = (__bf16)v;
    }
  }
}

// ---------------------------------------------------------------------------
// Kernel 3: flash-style attention with online softmax.
// Block: 8 waves x 16 q-rows = 128 q-rows per (b,h). k-tiles of 32.
// K tile staged via Tensor Data Mover (TENSOR_LOAD_TO_LDS + s_wait_tensorcnt);
// V tile staged cooperatively with an in-flight transpose.
// ---------------------------------------------------------------------------
__global__ __launch_bounds__(256) void attn_kernel(
    const __bf16* __restrict__ Qb, const __bf16* __restrict__ Kg,
    const __bf16* __restrict__ Vg,
    const float* __restrict__ aw, const unsigned char* __restrict__ mask,
    const float* __restrict__ geom, __bf16* __restrict__ Attn) {
  __shared__ __bf16 Kt[32][64];        // K tile (k-major), TDM destination
  __shared__ __bf16 Vt[64][32];        // V tile transposed (d-major)
  __shared__ float  Pst[8][32][16];    // per-wave P staging, [wave][k][q]

  const int bh = blockIdx.y;
  const int b = bh >> 3, h = bh & 7;
  const int wave = threadIdx.x >> 5;
  const int lane = threadIdx.x & 31;
  const int half = lane >> 4;
  const int cl   = lane & 15;
  const int q0 = blockIdx.x * 128 + wave * 16;

  const __bf16* Qbase = Qb + ((size_t)bh * NQ_ + q0) * DK_;
  const __bf16* Kbase = Kg + (size_t)bh * NK_ * DK_;
  const __bf16* Vbase = Vg + (size_t)bh * NK_ * DK_;
  const size_t  sbase = (size_t)bh * NQ_ * NK_;
  const unsigned kt_off = (unsigned)(uintptr_t)(&Kt[0][0]);

  const v16bf aq0 = load_frag_bf16(Qbase + 0, DK_);
  const v16bf aq1 = load_frag_bf16(Qbase + 32, DK_);

  v8f o[4] = {};
  float mrun[8], lrun[8];
#pragma unroll
  for (int g = 0; g < 8; ++g) { mrun[g] = -1e30f; lrun[g] = 0.f; }

  for (int k0 = 0; k0 < NK_; k0 += 32) {
    __syncthreads();
    {  // cooperative stage of V (transposed) into LDS
      const int t  = threadIdx.x;
      const int r  = t >> 3;             // 0..31 (k within tile)
      const int c8 = (t & 7) << 3;       // 0..56 (d chunk)
      uint4 vd = *(const uint4*)(Vbase + (size_t)(k0 + r) * DK_ + c8);
      const __bf16* ve = (const __bf16*)&vd;
#pragma unroll
      for (int i = 0; i < 8; ++i) Vt[c8 + i][r] = ve[i];
    }
    if (threadIdx.x < 32) {  // wave 0: DMA the K tile via the Tensor Data Mover
      tdm_load_2d_bf16(kt_off, Kbase + (size_t)k0 * DK_,
                       /*rows=*/32, /*row_elems=*/DK_,
                       /*row_stride=*/DK_, /*tensor_rows=*/NK_ - k0);
      __builtin_amdgcn_s_wait_tensorcnt((short)0);
    }
    __syncthreads();

    // scores S = Q * K^T / sqrt(DK): two 16x16 tiles over the 32 k-cols
    v8f s0 = {}, s1 = {};
    {
      v16bf b00 = load_frag_bf16(&Kt[0][0], 64);
      v16bf b01 = load_frag_bf16(&Kt[0][32], 64);
      s0 = wmma_acc(aq0, b00, s0);
      s0 = wmma_acc(aq1, b01, s0);
      v16bf b10 = load_frag_bf16(&Kt[16][0], 64);
      v16bf b11 = load_frag_bf16(&Kt[16][32], 64);
      s1 = wmma_acc(aq0, b10, s1);
      s1 = wmma_acc(aq1, b11, s1);
    }

    // elementwise: *aw, mask -> -inf, + log(clip(geometry))
    float tv0[8], tv1[8];
#pragma unroll
    for (int g = 0; g < 8; ++g) {
      const int q = q0 + g + (half << 3);
      const size_t rowi = sbase + (size_t)q * NK_ + (size_t)k0 + cl;
      const float a0 = aw[rowi];
      const float a1 = aw[rowi + 16];
      float t0 = mask[rowi]      ? -1e30f : s0[g] * 0.125f * a0;
      float t1 = mask[rowi + 16] ? -1e30f : s1[g] * 0.125f * a1;
      t0 += __logf(fmaxf(geom[rowi], 1e-6f));
      t1 += __logf(fmaxf(geom[rowi + 16], 1e-6f));
      tv0[g] = t0; tv1[g] = t1;
      if (k0 + 32 < NK_) {  // stream-prefetch next k-tile of the big tensors
        __builtin_prefetch(&aw[rowi + 32], 0, 0);
        __builtin_prefetch(&geom[rowi + 32], 0, 0);
      }
    }

    // online softmax: 16-lane butterfly reductions (wave32-safe)
#pragma unroll
    for (int g = 0; g < 8; ++g) {
      float mx = fmaxf(tv0[g], tv1[g]);
      mx = fmaxf(mx, __shfl_xor(mx, 1, 16));
      mx = fmaxf(mx, __shfl_xor(mx, 2, 16));
      mx = fmaxf(mx, __shfl_xor(mx, 4, 16));
      mx = fmaxf(mx, __shfl_xor(mx, 8, 16));
      const float mnew = fmaxf(mrun[g], mx);
      const float fs = __expf(mrun[g] - mnew);
      mrun[g] = mnew;
      const float e0 = __expf(tv0[g] - mnew);
      const float e1 = __expf(tv1[g] - mnew);
      float rs = e0 + e1;
      rs += __shfl_xor(rs, 1, 16);
      rs += __shfl_xor(rs, 2, 16);
      rs += __shfl_xor(rs, 4, 16);
      rs += __shfl_xor(rs, 8, 16);
      lrun[g] = lrun[g] * fs + rs;
      tv0[g] = e0; tv1[g] = e1;
#pragma unroll
      for (int t = 0; t < 4; ++t) o[t][g] = o[t][g] * fs;
    }

    // stage P (C-layout) into per-wave LDS column-major, read back as A-frag
    {
      float* p0 = &Pst[wave][cl][half << 3];
      *(float4*)p0       = make_float4(tv0[0], tv0[1], tv0[2], tv0[3]);
      *((float4*)p0 + 1) = make_float4(tv0[4], tv0[5], tv0[6], tv0[7]);
      float* p1 = &Pst[wave][16 + cl][half << 3];
      *(float4*)p1       = make_float4(tv1[0], tv1[1], tv1[2], tv1[3]);
      *((float4*)p1 + 1) = make_float4(tv1[4], tv1[5], tv1[6], tv1[7]);
    }
    v16bf pa;
    {
      union { v16bf v; __bf16 e[16]; } u;
      const int kh = half << 3;
#pragma unroll
      for (int i = 0; i < 8; ++i) u.e[i]     = (__bf16)Pst[wave][kh + i][cl];
#pragma unroll
      for (int i = 0; i < 8; ++i) u.e[8 + i] = (__bf16)Pst[wave][16 + kh + i][cl];
      pa = u.v;
    }

    // O += P * V   (B-fragments contiguous thanks to transposed Vt)
#pragma unroll
    for (int t = 0; t < 4; ++t) {
      v16bf bvf = load_frag_bf16(&Vt[16 * t][0], 32);
      o[t] = wmma_acc(pa, bvf, o[t]);
    }
  }

  // normalize and store bf16 [b][q][h*dk]
#pragma unroll
  for (int t = 0; t < 4; ++t) {
#pragma unroll
    for (int g = 0; g < 8; ++g) {
      const int q = q0 + g + (half << 3);
      const int d = t * 16 + cl;
      const float v = o[t][g] / fmaxf(lrun[g], 1e-30f);
      Attn[((size_t)b * NQ_ + q) * HD_ + h * DK_ + d] = (__bf16)v;
    }
  }
}

// ---------------------------------------------------------------------------
// Kernel 4: output projection (M=8192, N=512, K=512) + bias -> f32 out
// ---------------------------------------------------------------------------
__global__ __launch_bounds__(256) void outproj_kernel(
    const __bf16* __restrict__ Attn, const __bf16* __restrict__ WoT,
    const float* __restrict__ bo, float* __restrict__ out) {
  const int wave = threadIdx.x >> 5;
  const int lane = threadIdx.x & 31;
  const int half = lane >> 4;
  const int cl   = lane & 15;
  const int m0 = blockIdx.x * 128 + wave * 16;
  const int n0 = blockIdx.y * 128;

  v8f acc[8] = {};
  for (int k = 0; k < HD_; k += 32) {
    v16bf a = load_frag_bf16(Attn + (size_t)m0 * HD_ + k, HD_);
#pragma unroll
    for (int t = 0; t < 8; ++t) {
      v16bf b = load_frag_bf16(WoT + (size_t)(n0 + 16 * t) * HD_ + k, HD_);
      acc[t] = wmma_acc(a, b, acc[t]);
    }
  }
#pragma unroll
  for (int t = 0; t < 8; ++t) {
    const int col = n0 + t * 16 + cl;
    const float bc = bo[col];
#pragma unroll
    for (int g = 0; g < 8; ++g) {
      const int row = m0 + g + (half << 3);
      out[(size_t)row * DM_ + col] = acc[t][g] + bc;
    }
  }
}

// ---------------------------------------------------------------------------
extern "C" void kernel_launch(void* const* d_in, const int* in_sizes, int n_in,
                              void* d_out, int out_size, void* d_ws, size_t ws_size,
                              hipStream_t stream) {
  (void)in_sizes; (void)n_in; (void)out_size; (void)ws_size;
  const float* queries = (const float*)d_in[0];
  const float* keys    = (const float*)d_in[1];
  const float* values  = (const float*)d_in[2];
  const unsigned char* mask = (const unsigned char*)d_in[3];
  const float* aw      = (const float*)d_in[4];
  const float* memory  = (const float*)d_in[5];
  const float* geom    = (const float*)d_in[6];
  const float* Wq = (const float*)d_in[7];   const float* bq  = (const float*)d_in[8];
  const float* Wk = (const float*)d_in[9];   const float* bk  = (const float*)d_in[10];
  const float* Wv = (const float*)d_in[11];  const float* bv  = (const float*)d_in[12];
  const float* Wmm = (const float*)d_in[13]; const float* bmm = (const float*)d_in[14];
  const float* Wo = (const float*)d_in[15];  const float* bo  = (const float*)d_in[16];
  float* out = (float*)d_out;

  char* ws = (char*)d_ws;
  size_t off = 0;
  auto carve = [&](size_t bytes) {
    char* p = ws + off;
    off += (bytes + 255) & ~(size_t)255;
    return p;
  };
  __bf16* WqT  = (__bf16*)carve((size_t)DM_ * HD_ * 2);
  __bf16* WkT  = (__bf16*)carve((size_t)DM_ * HD_ * 2);
  __bf16* WvT  = (__bf16*)carve((size_t)DM_ * HD_ * 2);
  __bf16* WoT  = (__bf16*)carve((size_t)HD_ * DM_ * 2);
  __bf16* Qb   = (__bf16*)carve((size_t)B_ * H_ * NQ_ * DK_ * 2);
  __bf16* Kgp  = (__bf16*)carve((size_t)B_ * H_ * NK_ * DK_ * 2);
  __bf16* Vgp  = (__bf16*)carve((size_t)B_ * H_ * NK_ * DK_ * 2);
  __bf16* Attn = (__bf16*)carve((size_t)B_ * NQ_ * HD_ * 2);

  pack_weights_kernel<<<dim3(1024, 4), 256, 0, stream>>>(
      Wq, Wk, Wv, Wo, WqT, WkT, WvT, WoT);

  proj_kernel<<<dim3(64, 4, 3), 256, 0, stream>>>(
      queries, keys, values, WqT, WkT, WvT, bq, bk, bv,
      memory, Wmm, bmm, Qb, Kgp, Vgp);

  attn_kernel<<<dim3(NQ_ / 128, B_ * H_), 256, 0, stream>>>(
      Qb, Kgp, Vgp, aw, mask, geom, Attn);

  outproj_kernel<<<dim3(64, 4), 256, 0, stream>>>(Attn, WoT, bo, out);
}
